// CausalSelfAttention_59493886984270
// MI455X (gfx1250) — compile-verified, matrix operand order checked by reading, not run on hardware
//
#include <hip/hip_runtime.h>
#include <hip/hip_bf16.h>
#include <math.h>

// ---------------------------------------------------------------------------
// CDNA5 / gfx1250 wave32 WMMA implementation of causal GQA attention w/ ALiBi.
// All GEMMs (QKV proj, Q@K^T, P@V, out proj) run on v_wmma_f32_16x16x32_bf16.
// GEMM: 128x256 block tile, 64x64 wave tile (16 WMMAs / 8 frag loads),
// double-buffered LDS so global loads overlap the WMMA burst.
// ---------------------------------------------------------------------------

typedef __attribute__((ext_vector_type(16))) __bf16 v16bf;
typedef __attribute__((ext_vector_type(8)))  __bf16 v8bf;
typedef __attribute__((ext_vector_type(8)))  float  v8f;

union V16 { v16bf v; v8bf h[2]; };

static constexpr int Bd  = 2;
static constexpr int Td  = 2048;
static constexpr int Cd  = 2048;
static constexpr int Hd  = 16;
static constexpr int HKV = 4;
static constexpr int Dd  = 128;
static constexpr int KVC = 512;   // HKV * Dd

__device__ __forceinline__ v8f wmma_bf16(v16bf a, v16bf b, v8f c) {
  // D = A(16x32 bf16) x B(32x16 bf16) + C(16x16 f32)
  return __builtin_amdgcn_wmma_f32_16x16x32_bf16(
      /*neg_a=*/false, a, /*neg_b=*/false, b,
      /*c_mod=*/(short)0, c, /*reuse_a=*/false, /*reuse_b=*/false);
}

// Load a 16x32 A-style (or 32x16 B-style) bf16 fragment from an LDS tile whose
// rows are K-contiguous. Implements the ISA half-wave K swizzle:
//   lanes 0-15  hold K = {0..7, 16..23}
//   lanes 16-31 hold K = {8..15, 24..31}
__device__ __forceinline__ v16bf load_frag_lds(const __bf16* base, int pitch,
                                               int row, int lane) {
  const int half = (lane >> 4) & 1;
  const int r    = lane & 15;
  const __bf16* p = base + (row + r) * pitch + half * 8;
  V16 u;
  u.h[0] = *(const v8bf*)(p);
  u.h[1] = *(const v8bf*)(p + 16);
  return u.v;
}

// ---------------------------------------------------------------------------
// f32 -> bf16 conversion (grid-stride)
// ---------------------------------------------------------------------------
__global__ void cvt_f32_bf16(const float* __restrict__ in,
                             __bf16* __restrict__ out, int n) {
  int i = blockIdx.x * blockDim.x + threadIdx.x;
  int stride = gridDim.x * blockDim.x;
  for (; i < n; i += stride) out[i] = (__bf16)in[i];
}

// ---------------------------------------------------------------------------
// NT GEMM: C[M,N] = A[M,K] * W[N,K]^T   (both operands K-contiguous rows)
// Block 128x256, 8 waves, each wave 64x64 (4x4 16x16 fragments).
// Double-buffered LDS (ping-pong) hides global latency behind WMMAs.
// OMODE 0: f32 row-major out. 1: bf16 row-major out. 2: bf16 transposed out
//          ([b][n][t] with t = m % Tld) so V is stored as V^T for attention.
// ---------------------------------------------------------------------------
template <int OMODE>
__global__ __launch_bounds__(256, 2) void gemm_nt_bf16(
    const __bf16* __restrict__ A, const __bf16* __restrict__ W,
    void* __restrict__ outp, int M, int N, int K, int Tld) {
  constexpr int BM = 128, BN = 256, BK = 32, LP = 40;  // LP: padded LDS pitch
  __shared__ __bf16 As[2][BM * LP];   // 2 x 10 KB
  __shared__ __bf16 Bs[2][BN * LP];   // 2 x 20 KB

  const int tid  = threadIdx.x;
  const int lane = tid & 31;
  const int wid  = tid >> 5;          // 8 waves
  const int wm   = wid & 1;           // 2 waves along M (64 rows each)
  const int wn   = wid >> 1;          // 4 waves along N (64 cols each)
  const int m0   = blockIdx.y * BM;
  const int n0   = blockIdx.x * BN;

  // Per-thread staging registers for the global->LDS pipeline.
  const int arow = tid >> 2, acol = (tid & 3) << 3;  // A: 2 chunks/thread
  v8bf ar[2], br[4];                                 // B: 4 chunks/thread

  auto load_regs = [&](int kt) {
    const int k0 = kt * BK;
#pragma unroll
    for (int i = 0; i < 2; ++i) {
      const int c = tid + i * 256;
      ar[i] = *(const v8bf*)&A[(size_t)(m0 + (c >> 2)) * K + k0 + ((c & 3) << 3)];
    }
#pragma unroll
    for (int i = 0; i < 4; ++i) {
      const int c = tid + i * 256;
      br[i] = *(const v8bf*)&W[(size_t)(n0 + (c >> 2)) * K + k0 + ((c & 3) << 3)];
    }
  };
  auto store_lds = [&](int buf) {
#pragma unroll
    for (int i = 0; i < 2; ++i) {
      const int c = tid + i * 256;
      *(v8bf*)&As[buf][(c >> 2) * LP + ((c & 3) << 3)] = ar[i];
    }
#pragma unroll
    for (int i = 0; i < 4; ++i) {
      const int c = tid + i * 256;
      *(v8bf*)&Bs[buf][(c >> 2) * LP + ((c & 3) << 3)] = br[i];
    }
  };

  const v8f zero8 = {0.f, 0.f, 0.f, 0.f, 0.f, 0.f, 0.f, 0.f};
  v8f acc[4][4];
#pragma unroll
  for (int i = 0; i < 4; ++i)
#pragma unroll
    for (int j = 0; j < 4; ++j) acc[i][j] = zero8;

  const int nkt = K >> 5;  // K / BK
  load_regs(0);
  store_lds(0);
  __syncthreads();

  for (int kt = 0; kt < nkt; ++kt) {
    const int cur = kt & 1;
    if (kt + 1 < nkt) load_regs(kt + 1);  // global loads in flight over WMMAs

    v16bf af[4];
#pragma unroll
    for (int i = 0; i < 4; ++i)
      af[i] = load_frag_lds(As[cur], LP, wm * 64 + i * 16, lane);
#pragma unroll
    for (int j = 0; j < 4; ++j) {
      v16bf bf = load_frag_lds(Bs[cur], LP, wn * 64 + j * 16, lane);
#pragma unroll
      for (int i = 0; i < 4; ++i)
        acc[i][j] = wmma_bf16(af[i], bf, acc[i][j]);
    }

    if (kt + 1 < nkt) {
      store_lds(cur ^ 1);
      __syncthreads();
    }
  }

  // Epilogue. D layout: lanes 0-15 -> M=r, N=lane; lanes 16-31 -> M=8+r.
  const int half = lane >> 4;
  const int nl   = lane & 15;
#pragma unroll
  for (int i = 0; i < 4; ++i) {
#pragma unroll
    for (int j = 0; j < 4; ++j) {
      const int fm = m0 + wm * 64 + i * 16 + half * 8;  // rows fm..fm+7
      const int fn = n0 + wn * 64 + j * 16 + nl;
      if constexpr (OMODE == 0) {
        float* O = (float*)outp;
#pragma unroll
        for (int r = 0; r < 8; ++r)
          O[(size_t)(fm + r) * N + fn] = acc[i][j][r];
      } else if constexpr (OMODE == 1) {
        __bf16* O = (__bf16*)outp;
#pragma unroll
        for (int r = 0; r < 8; ++r)
          O[(size_t)(fm + r) * N + fn] = (__bf16)acc[i][j][r];
      } else {
        // transposed: out[b][fn][t], 8 consecutive t -> one b128 store
        const int b = fm / Tld;
        const int t = fm - b * Tld;
        __bf16* O = (__bf16*)outp + (size_t)b * N * Tld + (size_t)fn * Tld + t;
        v8bf pk;
#pragma unroll
        for (int r = 0; r < 8; ++r) pk[r] = (__bf16)acc[i][j][r];
        *(v8bf*)O = pk;
      }
    }
  }
}

// ---------------------------------------------------------------------------
// Flash attention with causal mask + ALiBi.
// Block = 128 threads (4 waves); each block: one (b,h), 64 query rows.
// Q: [B*T, C] bf16 (head slice), K: [B*T, KVC] bf16, Vt: [B][KVC][T] bf16.
// Y: [B*T, C] bf16 output.
// ---------------------------------------------------------------------------
__global__ __launch_bounds__(128, 2) void attn_flash(
    const __bf16* __restrict__ Q, const __bf16* __restrict__ Kg,
    const __bf16* __restrict__ Vt, __bf16* __restrict__ Y) {
  __shared__ __bf16 Ks[64 * 136];    // [key][d], padded pitch 136
  __shared__ __bf16 Vts[128 * 72];   // [d][key], padded pitch 72
  __shared__ __bf16 Ps[4][16 * 72];  // per-wave P staging [m][key]

  const int tid  = threadIdx.x;
  const int lane = tid & 31;
  const int w    = tid >> 5;  // 4 waves * 16 q rows = 64 q rows
  const int half = lane >> 4;
  const int nl   = lane & 15;

  const int qb = blockIdx.x;        // query block (64 rows)
  const int bh = blockIdx.y;
  const int b  = bh >> 4;
  const int h  = bh & 15;
  const int hkv = h >> 2;           // repeat_interleave(4)
  const int q0  = qb * 64;

  const float slope = exp2f(-0.5f * (float)(h + 1));  // ALiBi, H=16 pow2 path
  const float scale = 0.08838834764831845f;           // 1/sqrt(128)

  // Q fragments for this wave's 16 rows, all 128 d (4 K-steps), kept in VGPRs.
  const __bf16* Qrow =
      Q + (size_t)(b * Td + q0 + w * 16 + nl) * Cd + h * Dd;
  v16bf qf[4];
#pragma unroll
  for (int ks = 0; ks < 4; ++ks) {
    V16 u;
    u.h[0] = *(const v8bf*)(Qrow + ks * 32 + half * 8);
    u.h[1] = *(const v8bf*)(Qrow + ks * 32 + 16 + half * 8);
    qf[ks] = u.v;
  }

  const v8f zero8 = {0.f, 0.f, 0.f, 0.f, 0.f, 0.f, 0.f, 0.f};
  float mrow[8], lrow[8];
#pragma unroll
  for (int r = 0; r < 8; ++r) { mrow[r] = -1e30f; lrow[r] = 0.f; }
  v8f o[8];
#pragma unroll
  for (int j2 = 0; j2 < 8; ++j2) o[j2] = zero8;

  const __bf16* Kbase  = Kg + (size_t)b * Td * KVC + hkv * Dd;
  const __bf16* Vtbase = Vt + ((size_t)b * KVC + hkv * Dd) * Td;

  for (int kb = 0; kb <= qb; ++kb) {
    __syncthreads();
    // Stage K tile [64 keys][128 d] and Vt tile [128 d][64 keys].
#pragma unroll
    for (int i = 0; i < 8; ++i) {
      int c = tid + i * 128;
      int row = c >> 4, col = (c & 15) << 3;
      *(v8bf*)&Ks[row * 136 + col] =
          *(const v8bf*)&Kbase[(size_t)(kb * 64 + row) * KVC + col];
    }
#pragma unroll
    for (int i = 0; i < 8; ++i) {
      int c = tid + i * 128;
      int row = c >> 3, col = (c & 7) << 3;
      *(v8bf*)&Vts[row * 72 + col] =
          *(const v8bf*)&Vtbase[(size_t)row * Td + kb * 64 + col];
    }
    __syncthreads();

    // S = Q @ K^T  (16 q rows x 64 keys) : 16 WMMAs
    v8f s[4] = {zero8, zero8, zero8, zero8};
#pragma unroll
    for (int ks = 0; ks < 4; ++ks) {
#pragma unroll
      for (int j = 0; j < 4; ++j) {
        v16bf kf = load_frag_lds(Ks + ks * 32, 136, j * 16, lane);
        s[j] = wmma_bf16(qf[ks], kf, s[j]);
      }
    }

    // Scale, clamp, causal + ALiBi, online softmax (row = r + 8*half across
    // one 16-lane half -> shfl_xor(width 16) row reductions).
    float alpha[8];
#pragma unroll
    for (int r = 0; r < 8; ++r) {
      const int qi = q0 + w * 16 + r + half * 8;
      float mx = -1e30f;
#pragma unroll
      for (int j = 0; j < 4; ++j) {
        const int kj = kb * 64 + j * 16 + nl;
        float v = s[j][r] * scale;
        v = fminf(fmaxf(v, -10000.f), 10000.f);
        v = (kj > qi) ? -1e30f : (v + slope * (float)(kj - qi));
        s[j][r] = v;
        mx = fmaxf(mx, v);
      }
#pragma unroll
      for (int d1 = 1; d1 < 16; d1 <<= 1) mx = fmaxf(mx, __shfl_xor(mx, d1, 16));
      const float mnew = fmaxf(mrow[r], mx);
      alpha[r] = __expf(mrow[r] - mnew);
      float rs = 0.f;
#pragma unroll
      for (int j = 0; j < 4; ++j) {
        const float p = __expf(s[j][r] - mnew);
        s[j][r] = p;
        rs += p;
      }
#pragma unroll
      for (int d1 = 1; d1 < 16; d1 <<= 1) rs += __shfl_xor(rs, d1, 16);
      lrow[r] = lrow[r] * alpha[r] + rs;
      mrow[r] = mnew;
    }

    // Rescale O, stage P (bf16, A-fragment-readable [m][key] layout).
#pragma unroll
    for (int j2 = 0; j2 < 8; ++j2)
#pragma unroll
      for (int r = 0; r < 8; ++r) o[j2][r] *= alpha[r];
#pragma unroll
    for (int j = 0; j < 4; ++j)
#pragma unroll
      for (int r = 0; r < 8; ++r)
        Ps[w][(r + half * 8) * 72 + j * 16 + nl] = (__bf16)s[j][r];

    // O += P @ V : 16 WMMAs (K = 64 keys = 2 steps, N = 128 dims = 8 tiles)
#pragma unroll
    for (int ks2 = 0; ks2 < 2; ++ks2) {
      v16bf pf = load_frag_lds(&Ps[w][0] + ks2 * 32, 72, 0, lane);
#pragma unroll
      for (int j2 = 0; j2 < 8; ++j2) {
        v16bf vf = load_frag_lds(Vts + ks2 * 32, 72, j2 * 16, lane);
        o[j2] = wmma_bf16(pf, vf, o[j2]);
      }
    }
  }

  // Normalize and store Y[b*T + q, h*D + d] as bf16.
  float inv[8];
#pragma unroll
  for (int r = 0; r < 8; ++r) inv[r] = 1.0f / lrow[r];
#pragma unroll
  for (int j2 = 0; j2 < 8; ++j2) {
#pragma unroll
    for (int r = 0; r < 8; ++r) {
      const int row = b * Td + q0 + w * 16 + r + half * 8;
      const int col = h * Dd + j2 * 16 + nl;
      Y[(size_t)row * Cd + col] = (__bf16)(o[j2][r] * inv[r]);
    }
  }
}

// ---------------------------------------------------------------------------
// Launch
// ---------------------------------------------------------------------------
extern "C" void kernel_launch(void* const* d_in, const int* in_sizes, int n_in,
                              void* d_out, int out_size, void* d_ws,
                              size_t ws_size, hipStream_t stream) {
  const float* x  = (const float*)d_in[0];
  const float* qw = (const float*)d_in[1];
  const float* kw = (const float*)d_in[2];
  const float* vw = (const float*)d_in[3];
  const float* ow = (const float*)d_in[4];

  char* ws = (char*)d_ws;
  __bf16* xb  = (__bf16*)(ws + 0);          // 16 MB  [B*T, C]
  __bf16* qwb = (__bf16*)(ws + 16777216);   //  8 MB  [C, C]
  __bf16* kwb = (__bf16*)(ws + 25165824);   //  2 MB  [KVC, C]
  __bf16* vwb = (__bf16*)(ws + 27262976);   //  2 MB  [KVC, C]
  __bf16* owb = (__bf16*)(ws + 29360128);   //  8 MB  [C, C]
  __bf16* Qb  = (__bf16*)(ws + 37748736);   // 16 MB  [B*T, C]
  __bf16* Kb  = (__bf16*)(ws + 54525952);   //  4 MB  [B*T, KVC]
  __bf16* Vtb = (__bf16*)(ws + 58720256);   //  4 MB  [B][KVC][T]
  __bf16* Yb  = (__bf16*)(ws + 62914560);   // 16 MB  [B*T, C]

  // fp32 -> bf16 staging
  cvt_f32_bf16<<<1024, 256, 0, stream>>>(x,  xb,  Bd * Td * Cd);
  cvt_f32_bf16<<<512,  256, 0, stream>>>(qw, qwb, Cd * Cd);
  cvt_f32_bf16<<<256,  256, 0, stream>>>(kw, kwb, KVC * Cd);
  cvt_f32_bf16<<<256,  256, 0, stream>>>(vw, vwb, KVC * Cd);
  cvt_f32_bf16<<<512,  256, 0, stream>>>(ow, owb, Cd * Cd);

  const int M = Bd * Td;  // 4096
  dim3 gq(Cd / 256, M / 128);   // (8, 32)
  dim3 gk(KVC / 256, M / 128);  // (2, 32)

  gemm_nt_bf16<1><<<gq, 256, 0, stream>>>(xb, qwb, Qb, M, Cd, Cd, 1);
  gemm_nt_bf16<1><<<gk, 256, 0, stream>>>(xb, kwb, Kb, M, KVC, Cd, 1);
  gemm_nt_bf16<2><<<gk, 256, 0, stream>>>(xb, vwb, Vtb, M, KVC, Cd, Td);

  attn_flash<<<dim3(Td / 64, Bd * Hd), 128, 0, stream>>>(Qb, Kb, Vtb, Yb);

  gemm_nt_bf16<0><<<gq, 256, 0, stream>>>(Yb, owb, d_out, M, Cd, Cd, 1);
}
